// GINEClassifier_56221121904785
// MI455X (gfx1250) — matrix operand order repaired and checked by reading, not compiled, not run on hardware
//
#include <hip/hip_runtime.h>
#include <hip/hip_bf16.h>
#include <math.h>

// ---------------------------------------------------------------- constants
#define B_    64
#define N_    2048
#define E_    8192
#define FN_   34
#define FHC_  193
#define L_    5
#define H_    128
#define NC_   9
#define JK_   768
#define FUS_  256
#define EPS_  1e-5f
#define R_    (B_ * N_)          // 131072 rows
#define MT_   4                  // M subtiles (16 rows each) per block
#define BM_   (16 * MT_)         // 64 rows per block

typedef __attribute__((ext_vector_type(16))) __bf16 bf16x16;
typedef __attribute__((ext_vector_type(8)))  float  f32x8;

// ------------------------------------------------------------- input layout
enum {
  IN_NODE = 0, IN_EIDX, IN_ETYPE, IN_MASK, IN_HC,
  P_ENC_W, P_ENC_B, P_ENC_BN_G, P_ENC_BN_B, P_EDGE_EMB, P_GINE_EPS,
  P_GINE_W1, P_GINE_B1, P_GINE_BN1_G, P_GINE_BN1_B,
  P_GINE_W2, P_GINE_B2, P_GINE_BN2_G, P_GINE_BN2_B,
  P_LN_G, P_LN_B, P_VN_INIT,
  P_VN_W1, P_VN_B1, P_VN_BN1_G, P_VN_BN1_B,
  P_VN_W2, P_VN_B2, P_VN_BN2_G, P_VN_BN2_B,
  P_VN_GATE, P_ATTN_W, P_ATTN_WB, P_ATTN_A, P_ATTN_GATE,
  P_GP_W, P_GP_B, P_GP_BN_G, P_GP_BN_B,
  P_FE_W1, P_FE_B1, P_FE_BN1_G, P_FE_BN1_B,
  P_FE_W2, P_FE_B2, P_FE_BN2_G, P_FE_BN2_B,
  P_CLS_W1, P_CLS_B1, P_CLS_BN_G, P_CLS_BN_B,
  P_CLS_W2, P_CLS_B2
};

// =========================================================== WMMA bf16 GEMM
// C[M x 128] = A[M x K](bf16, row stride lda) @ Wt^T (+bias). Wt is N x K
// row-major bf16 (W pre-transposed).  Block: 256 thr = 8 waves; wave w owns
// output columns [16w,16w+16); block owns BM_=64 rows of A, staged in LDS
// (padded rows, 4-bank skew) and register-blocked 4x against one B fragment.
// Epilogues:
//   stats != 0 : atomically accumulate per-column sum/sumsq (fused BN stats)
//   scores!= 0 : attention mode — scores[row] += sum_col tanh(val)*a[col],
//                cross-lane butterfly reduce, no C write.
__global__ __launch_bounds__(256)
void k_gemm_bf16(const __bf16* __restrict__ A, int lda,
                 const __bf16* __restrict__ Wt,
                 const float* __restrict__ bias,
                 float* __restrict__ C, int ldc, int M, int K,
                 float* __restrict__ stats,
                 const float* __restrict__ attn_a,
                 float* __restrict__ scores)
{
  extern __shared__ __align__(16) char smem[];
  __bf16* As = (__bf16*)smem;                 // BM_ rows, stride K+8 halves
  const int ldsK = K + 8;                     // +16B pad per row (bank skew)

  const int tid  = threadIdx.x;
  const int wave = tid >> 5;
  const int lane = tid & 31;
  const int grp  = lane >> 4;                 // lane half (0/1)
  const int l16  = lane & 15;
  const int mbase = blockIdx.x * BM_;

  // ---- cooperative stage of A tile (BM_ x K) into LDS (dword granular)
  {
    unsigned* dst = (unsigned*)As;
    const int rowDW  = K / 2;                 // dwords per source row
    const int ldsDW  = ldsK / 2;              // dwords per padded row
    for (int i = tid; i < BM_ * rowDW; i += 256) {
      int r = i / rowDW, c = i % rowDW;
      dst[r * ldsDW + c] =
          *reinterpret_cast<const unsigned*>(A + (size_t)(mbase + r) * lda + 2 * c);
    }
  }
  __syncthreads();

  const int ncol = wave * 16 + l16;
  const __bf16* Wrow = Wt + (size_t)ncol * K;

  f32x8 acc[MT_];
  #pragma unroll
  for (int mt = 0; mt < MT_; ++mt)
    acc[mt] = (f32x8){0.f, 0.f, 0.f, 0.f, 0.f, 0.f, 0.f, 0.f};

  union Frag { bf16x16 v; uint4 q[2]; } a, b;

  for (int k0 = 0; k0 < K; k0 += 32) {
    // B 32x16 fragment: halves k0+8g..+7 and k0+16+8g..+7 (matches A)
    b.q[0] = *reinterpret_cast<const uint4*>(Wrow + k0 + 8 * grp);
    b.q[1] = *reinterpret_cast<const uint4*>(Wrow + k0 + 16 + 8 * grp);
    #pragma unroll
    for (int mt = 0; mt < MT_; ++mt) {
      const __bf16* arow = As + (size_t)(mt * 16 + l16) * ldsK;
      // A 16x32 fragment: halves k0+8g..+7 and k0+16+8g..+7
      a.q[0] = *reinterpret_cast<const uint4*>(arow + k0 + 8 * grp);
      a.q[1] = *reinterpret_cast<const uint4*>(arow + k0 + 16 + 8 * grp);
      acc[mt] = __builtin_amdgcn_wmma_f32_16x16x32_bf16(false, a.v, false, b.v,
                                                        (short)0, acc[mt],
                                                        false, false);
    }
  }

  const float bv = bias ? bias[ncol] : 0.f;

  if (scores) {            // ---- fused attention epilogue (no C write)
    const float av = attn_a[ncol];
    #pragma unroll
    for (int mt = 0; mt < MT_; ++mt) {
      #pragma unroll
      for (int r = 0; r < 8; ++r) {
        int row = mbase + mt * 16 + grp * 8 + r;
        float val = tanhf(acc[mt][r] + bv) * av;
        val += __shfl_xor(val, 8, 32);
        val += __shfl_xor(val, 4, 32);
        val += __shfl_xor(val, 2, 32);
        val += __shfl_xor(val, 1, 32);
        if (l16 == 0) atomicAdd(&scores[row], val);
      }
    }
  } else {                 // ---- store + fused BN column stats
    float s = 0.f, s2 = 0.f;
    #pragma unroll
    for (int mt = 0; mt < MT_; ++mt) {
      #pragma unroll
      for (int r = 0; r < 8; ++r) {
        int row = mbase + mt * 16 + grp * 8 + r;
        float val = acc[mt][r] + bv;
        if (C && row < M) C[(size_t)row * ldc + ncol] = val;
        s += val; s2 += val * val;
      }
    }
    if (stats) {
      atomicAdd(&stats[ncol], s);
      atomicAdd(&stats[H_ + ncol], s2);
    }
  }
}

// ================================================================= utilities
__global__ void k_zero(float* __restrict__ p, long long n) {
  long long i = (long long)blockIdx.x * blockDim.x + threadIdx.x;
  if (i < n) p[i] = 0.f;
}

__global__ void k_add(const float* __restrict__ x, const float* __restrict__ y,
                      float* __restrict__ o, int n) {
  int i = blockIdx.x * blockDim.x + threadIdx.x;
  if (i < n) o[i] = x[i] + y[i];
}

// encoder: Y[R x H] = NF[R x 34] @ W[34 x H] + b
__global__ __launch_bounds__(256)
void k_enc_mm(const float* __restrict__ NF, const float* __restrict__ W,
              const float* __restrict__ bias, float* __restrict__ Y) {
  int idx = blockIdx.x * blockDim.x + threadIdx.x;
  if (idx >= R_ * H_) return;
  int row = idx / H_, col = idx % H_;
  const float* x = NF + (size_t)row * FN_;
  float s = bias[col];
  #pragma unroll
  for (int k = 0; k < FN_; ++k) s += x[k] * W[k * H_ + col];
  Y[idx] = s;
}

// per-column sum/sumsq over rows (H_=128 columns) — used for encoder only
#define CS_CHUNK 512
__global__ __launch_bounds__(128)
void k_colstats(const float* __restrict__ X, int rows, float* __restrict__ st) {
  int col = threadIdx.x;
  int r0 = blockIdx.x * CS_CHUNK;
  int r1 = r0 + CS_CHUNK; if (r1 > rows) r1 = rows;
  float s = 0.f, s2 = 0.f;
  for (int r = r0; r < r1; ++r) {
    float v = X[(size_t)r * H_ + col];
    s += v; s2 += v * v;
  }
  atomicAdd(&st[col], s);
  atomicAdd(&st[H_ + col], s2);
}

// BN apply (+optional relu); optional f32 out and/or bf16 out (JK stripe)
__global__ __launch_bounds__(256)
void k_bn_apply(const float* __restrict__ X, int rows,
                const float* __restrict__ st,
                const float* __restrict__ g, const float* __restrict__ b,
                int do_relu, float* __restrict__ outF,
                __bf16* __restrict__ outB, int ob_ld, int ob_off) {
  int idx = blockIdx.x * blockDim.x + threadIdx.x;
  if (idx >= rows * H_) return;
  int row = idx / H_, col = idx % H_;
  float inv = 1.f / (float)rows;
  float m = st[col] * inv;
  float var = st[H_ + col] * inv - m * m;
  float y = (X[idx] - m) * rsqrtf(var + EPS_) * g[col] + b[col];
  if (do_relu) y = fmaxf(y, 0.f);
  if (outF) outF[idx] = y;
  if (outB) outB[(size_t)row * ob_ld + ob_off + col] = (__bf16)y;
}

// message + scatter-add: agg[b,dst] += relu(h[b,src] + emb[etype])
__global__ __launch_bounds__(256)
void k_msg(const float* __restrict__ h, const int* __restrict__ eidx,
           const int* __restrict__ etype, const float* __restrict__ emb,
           float* __restrict__ agg) {
  long long t = (long long)blockIdx.x * blockDim.x + threadIdx.x;
  if (t >= (long long)B_ * E_ * (H_ / 4)) return;
  int c4 = (int)(t & 31) * 4;
  long long eg = t >> 5;
  int b = (int)(eg / E_), e = (int)(eg % E_);
  int src = eidx[(size_t)b * 2 * E_ + e];
  int dst = eidx[(size_t)b * 2 * E_ + E_ + e];
  int et  = etype[(size_t)b * E_ + e];
  const float4 hv = *reinterpret_cast<const float4*>(h + ((size_t)b * N_ + src) * H_ + c4);
  const float4 ev = *reinterpret_cast<const float4*>(emb + (size_t)et * H_ + c4);
  float* o = agg + ((size_t)b * N_ + dst) * H_ + c4;
  atomicAdd(o + 0, fmaxf(hv.x + ev.x, 0.f));
  atomicAdd(o + 1, fmaxf(hv.y + ev.y, 0.f));
  atomicAdd(o + 2, fmaxf(hv.z + ev.z, 0.f));
  atomicAdd(o + 3, fmaxf(hv.w + ev.w, 0.f));
}

// t2 = (1+eps_l)*h + agg  (+ bf16 copy for GEMM A)
__global__ __launch_bounds__(256)
void k_hnew(const float* __restrict__ h, const float* __restrict__ agg,
            const float* __restrict__ eps, int l,
            float* __restrict__ t2, __bf16* __restrict__ abf) {
  int idx = blockIdx.x * blockDim.x + threadIdx.x;
  if (idx >= R_ * H_) return;
  float v = (1.f + eps[l]) * h[idx] + agg[idx];
  t2[idx] = v;
  abf[idx] = (__bf16)v;
}

// pack W[K x N] (f32) -> Wt[N x K] (bf16)
__global__ __launch_bounds__(256)
void k_pack_w(const float* __restrict__ W, int K, int Ncols,
              __bf16* __restrict__ Wt) {
  int idx = blockIdx.x * blockDim.x + threadIdx.x;
  if (idx >= K * Ncols) return;
  int k = idx / Ncols, n = idx % Ncols;
  Wt[(size_t)n * K + k] = (__bf16)W[idx];
}

// LayerNorm of (h + d) per row, write into out (in-place on h is safe)
__global__ __launch_bounds__(128)
void k_ln_res(const float* __restrict__ h, const float* __restrict__ d,
              const float* __restrict__ g, const float* __restrict__ b,
              float* __restrict__ out) {
  __shared__ float red[128];
  int row = blockIdx.x, c = threadIdx.x;
  float v = h[(size_t)row * H_ + c] + d[(size_t)row * H_ + c];
  red[c] = v; __syncthreads();
  for (int s = 64; s > 0; s >>= 1) { if (c < s) red[c] += red[c + s]; __syncthreads(); }
  float mean = red[0] * (1.f / H_);
  __syncthreads();
  float dv = v - mean;
  red[c] = dv * dv; __syncthreads();
  for (int s = 64; s > 0; s >>= 1) { if (c < s) red[c] += red[c + s]; __syncthreads(); }
  float var = red[0] * (1.f / H_);
  out[(size_t)row * H_ + c] = dv * rsqrtf(var + EPS_) * g[c] + b[c];
}

// node_sum[b,c] = sum_n h[b,n,c]
__global__ __launch_bounds__(128)
void k_nodesum(const float* __restrict__ h, float* __restrict__ ns) {
  int b = blockIdx.x, c = threadIdx.x;
  float s = 0.f;
  for (int n = 0; n < N_; ++n) s += h[((size_t)b * N_ + n) * H_ + c];
  ns[b * H_ + c] = s;
}

__global__ void k_vn_init(const float* __restrict__ vi, float* __restrict__ vn) {
  int i = blockIdx.x * blockDim.x + threadIdx.x;
  if (i < B_ * H_) vn[i] = vi[i % H_];
}

// small dense GEMM (B<=64 rows) — scalar path, negligible work
__global__ void k_small_mm(const float* __restrict__ X, int K,
                           const float* __restrict__ W, const float* __restrict__ bias,
                           float* __restrict__ Y, int P, int rows) {
  int idx = blockIdx.x * blockDim.x + threadIdx.x;
  if (idx >= rows * P) return;
  int r = idx / P, p = idx % P;
  float s = bias ? bias[p] : 0.f;
  for (int k = 0; k < K; ++k) s += X[(size_t)r * K + k] * W[(size_t)k * P + p];
  Y[idx] = s;
}

__global__ void k_small_mm_cat(const float* __restrict__ X1, int K1,
                               const float* __restrict__ X2, int K2,
                               const float* __restrict__ W, const float* __restrict__ bias,
                               float* __restrict__ Y, int P, int rows) {
  int idx = blockIdx.x * blockDim.x + threadIdx.x;
  if (idx >= rows * P) return;
  int r = idx / P, p = idx % P;
  float s = bias ? bias[p] : 0.f;
  for (int k = 0; k < K1; ++k) s += X1[(size_t)r * K1 + k] * W[(size_t)k * P + p];
  for (int k = 0; k < K2; ++k) s += X2[(size_t)r * K2 + k] * W[(size_t)(K1 + k) * P + p];
  Y[idx] = s;
}

// in-place BN over `rows` rows, one thread per column
__global__ void k_small_bn(float* __restrict__ X, int rows, int P,
                           const float* __restrict__ g, const float* __restrict__ b,
                           int relu) {
  int p = blockIdx.x * blockDim.x + threadIdx.x;
  if (p >= P) return;
  float s = 0.f, s2 = 0.f;
  for (int r = 0; r < rows; ++r) { float v = X[(size_t)r * P + p]; s += v; s2 += v * v; }
  float m = s / rows, var = s2 / rows - m * m, inv = rsqrtf(var + EPS_);
  for (int r = 0; r < rows; ++r) {
    float y = (X[(size_t)r * P + p] - m) * inv * g[p] + b[p];
    if (relu) y = fmaxf(y, 0.f);
    X[(size_t)r * P + p] = y;
  }
}

// h += sigmoid(gate_l) * vn[b]; also emit bf16 JK stripe at coloff
__global__ __launch_bounds__(256)
void k_h_vn(float* __restrict__ h, const float* __restrict__ vn,
            const float* __restrict__ gatep, int l,
            __bf16* __restrict__ jk, int coloff) {
  int idx = blockIdx.x * blockDim.x + threadIdx.x;
  if (idx >= R_ * H_) return;
  int row = idx / H_, c = idx % H_;
  int b = row / N_;
  float gate = 1.f / (1.f + expf(-gatep[l]));
  float v = h[idx] + gate * vn[b * H_ + c];
  h[idx] = v;
  jk[(size_t)row * JK_ + coloff + c] = (__bf16)v;
}

// per-batch softmax over N + gated attn/sum pooling over JK channels
__global__ __launch_bounds__(256)
void k_pool(const float* __restrict__ sc, const __bf16* __restrict__ jk,
            const float* __restrict__ gatep, float* __restrict__ alpha,
            float* __restrict__ graw) {
  __shared__ float red[256];
  int b = blockIdx.x, t = threadIdx.x;
  const float* s = sc + (size_t)b * N_;
  float mx = -3.4e38f;
  for (int n = t; n < N_; n += 256) mx = fmaxf(mx, s[n]);
  red[t] = mx; __syncthreads();
  for (int k = 128; k > 0; k >>= 1) { if (t < k) red[t] = fmaxf(red[t], red[t + k]); __syncthreads(); }
  mx = red[0]; __syncthreads();
  float se = 0.f;
  for (int n = t; n < N_; n += 256) {
    float e = expf(s[n] - mx);
    alpha[(size_t)b * N_ + n] = e;
    se += e;
  }
  red[t] = se; __syncthreads();
  for (int k = 128; k > 0; k >>= 1) { if (t < k) red[t] += red[t + k]; __syncthreads(); }
  se = red[0];
  float gate = 1.f / (1.f + expf(-gatep[0]));
  const __bf16* J = jk + (size_t)b * N_ * JK_;
  const float* al = alpha + (size_t)b * N_;
  for (int c = t; c < JK_; c += 256) {
    float ap = 0.f, sp = 0.f;
    for (int n = 0; n < N_; ++n) {
      float jv = (float)J[(size_t)n * JK_ + c];
      ap += al[n] * jv;
      sp += jv;
    }
    graw[(size_t)b * JK_ + c] = gate * (ap / se) + (1.f - gate) * sp;
  }
}

// ================================================================== launcher
static inline dim3 g1(long long n, int bs) { return dim3((unsigned)((n + bs - 1) / bs)); }
static inline size_t gemm_lds(int K) { return (size_t)BM_ * (K / 2 + 4) * 4; }

extern "C" void kernel_launch(void* const* d_in, const int* in_sizes, int n_in,
                              void* d_out, int out_size, void* d_ws, size_t ws_size,
                              hipStream_t stream) {
  (void)in_sizes; (void)n_in; (void)out_size; (void)ws_size;

  const float* nf     = (const float*)d_in[IN_NODE];
  const int*   eidx   = (const int*)  d_in[IN_EIDX];
  const int*   etype  = (const int*)  d_in[IN_ETYPE];
  const float* hc     = (const float*)d_in[IN_HC];
  auto P = [&](int i) { return (const float*)d_in[i]; };

  // ---- workspace carve-up (~418 MB)
  char* w = (char*)d_ws;
  size_t off = 0;
  auto alloc = [&](size_t bytes) { size_t o = off; off = (off + bytes + 255) & ~(size_t)255; return o; };
  const size_t SZ_RH = (size_t)R_ * H_ * sizeof(float);        // 64 MB
  float*  h      = (float*) (w + alloc(SZ_RH));
  float*  t1     = (float*) (w + alloc(SZ_RH));
  float*  t2     = (float*) (w + alloc(SZ_RH));
  __bf16* abf    = (__bf16*)(w + alloc((size_t)R_ * H_ * 2));  // 32 MB
  __bf16* jk     = (__bf16*)(w + alloc((size_t)R_ * JK_ * 2)); // 192 MB
  __bf16* wbf    = (__bf16*)(w + alloc((size_t)JK_ * H_ * 2));
  float*  stats  = (float*) (w + alloc(2 * H_ * sizeof(float)));
  float*  scores = (float*) (w + alloc((size_t)R_ * sizeof(float)));
  float*  alpha  = (float*) (w + alloc((size_t)R_ * sizeof(float)));
  float*  nsum   = (float*) (w + alloc(B_ * H_ * sizeof(float)));
  float*  vn     = (float*) (w + alloc(B_ * H_ * sizeof(float)));
  float*  v0     = (float*) (w + alloc(B_ * H_ * sizeof(float)));
  float*  v1     = (float*) (w + alloc(B_ * H_ * sizeof(float)));
  float*  v2     = (float*) (w + alloc(B_ * H_ * sizeof(float)));
  float*  graw   = (float*) (w + alloc(B_ * JK_ * sizeof(float)));
  float*  graphL = (float*) (w + alloc(B_ * FUS_ * sizeof(float)));
  float*  f1     = (float*) (w + alloc(B_ * FUS_ * sizeof(float)));
  float*  f2     = (float*) (w + alloc(B_ * FUS_ * sizeof(float)));
  float*  xcls   = (float*) (w + alloc(B_ * H_ * sizeof(float)));

  const long long RH = (long long)R_ * H_;

  // ---------------- encoder: h = relu(bn(nf @ enc_W + enc_b)); JK stripe 0
  k_enc_mm<<<g1(RH, 256), 256, 0, stream>>>(nf, P(P_ENC_W), P(P_ENC_B), t1);
  k_zero<<<g1(2 * H_, 256), 256, 0, stream>>>(stats, 2 * H_);
  k_colstats<<<g1(R_, CS_CHUNK), 128, 0, stream>>>(t1, R_, stats);
  k_bn_apply<<<g1(RH, 256), 256, 0, stream>>>(t1, R_, stats, P(P_ENC_BN_G), P(P_ENC_BN_B),
                                              1, h, jk, JK_, 0);
  k_vn_init<<<g1(B_ * H_, 256), 256, 0, stream>>>(P(P_VN_INIT), vn);

  // ---------------- GINE layers
  for (int l = 0; l < L_; ++l) {
    const float* W1 = P(P_GINE_W1) + (size_t)l * H_ * H_;
    const float* W2 = P(P_GINE_W2) + (size_t)l * H_ * H_;

    // aggregate messages
    k_zero<<<g1(RH, 256), 256, 0, stream>>>(t1, RH);
    k_msg<<<g1((long long)B_ * E_ * 32, 256), 256, 0, stream>>>(h, eidx, etype, P(P_EDGE_EMB), t1);
    k_hnew<<<g1(RH, 256), 256, 0, stream>>>(h, t1, P(P_GINE_EPS), l, t2, abf);

    // MLP layer 1 (WMMA, fused BN stats) + BN + relu -> abf
    k_pack_w<<<g1(H_ * H_, 256), 256, 0, stream>>>(W1, H_, H_, wbf);
    k_zero<<<g1(2 * H_, 256), 256, 0, stream>>>(stats, 2 * H_);
    k_gemm_bf16<<<R_ / BM_, 256, gemm_lds(H_), stream>>>(
        abf, H_, wbf, P(P_GINE_B1) + l * H_, t1, H_, R_, H_,
        stats, nullptr, nullptr);
    k_bn_apply<<<g1(RH, 256), 256, 0, stream>>>(t1, R_, stats, P(P_GINE_BN1_G) + l * H_,
                                                P(P_GINE_BN1_B) + l * H_, 1,
                                                nullptr, abf, H_, 0);

    // MLP layer 2 (WMMA, fused BN stats) + BN -> t2 (f32)
    k_pack_w<<<g1(H_ * H_, 256), 256, 0, stream>>>(W2, H_, H_, wbf);
    k_zero<<<g1(2 * H_, 256), 256, 0, stream>>>(stats, 2 * H_);
    k_gemm_bf16<<<R_ / BM_, 256, gemm_lds(H_), stream>>>(
        abf, H_, wbf, P(P_GINE_B2) + l * H_, t2, H_, R_, H_,
        stats, nullptr, nullptr);
    k_bn_apply<<<g1(RH, 256), 256, 0, stream>>>(t2, R_, stats, P(P_GINE_BN2_G) + l * H_,
                                                P(P_GINE_BN2_B) + l * H_, 0,
                                                t2, nullptr, 0, 0);

    // residual + LayerNorm (in-place on h)
    k_ln_res<<<R_, 128, 0, stream>>>(h, t2, P(P_LN_G) + l * H_, P(P_LN_B) + l * H_, h);

    // virtual node
    k_nodesum<<<B_, 128, 0, stream>>>(h, nsum);
    k_add<<<g1(B_ * H_, 256), 256, 0, stream>>>(vn, nsum, v0, B_ * H_);
    k_small_mm<<<g1(B_ * H_, 256), 256, 0, stream>>>(v0, H_, P(P_VN_W1) + (size_t)l * H_ * H_,
                                                     P(P_VN_B1) + l * H_, v1, H_, B_);
    k_small_bn<<<g1(H_, 128), 128, 0, stream>>>(v1, B_, H_, P(P_VN_BN1_G) + l * H_,
                                                P(P_VN_BN1_B) + l * H_, 1);
    k_small_mm<<<g1(B_ * H_, 256), 256, 0, stream>>>(v1, H_, P(P_VN_W2) + (size_t)l * H_ * H_,
                                                     P(P_VN_B2) + l * H_, v2, H_, B_);
    k_small_bn<<<g1(H_, 128), 128, 0, stream>>>(v2, B_, H_, P(P_VN_BN2_G) + l * H_,
                                                P(P_VN_BN2_B) + l * H_, 0);
    k_add<<<g1(B_ * H_, 256), 256, 0, stream>>>(v2, vn, vn, B_ * H_);
    k_h_vn<<<g1(RH, 256), 256, 0, stream>>>(h, vn, P(P_VN_GATE), l, jk, (l + 1) * H_);
  }

  // ---------------- attention over JK (WMMA, K=768, fused tanh-dot epilogue)
  k_pack_w<<<g1(JK_ * H_, 256), 256, 0, stream>>>(P(P_ATTN_W), JK_, H_, wbf);
  k_zero<<<g1(R_, 256), 256, 0, stream>>>(scores, R_);
  k_gemm_bf16<<<R_ / BM_, 256, gemm_lds(JK_), stream>>>(
      jk, JK_, wbf, P(P_ATTN_WB), nullptr, 0, R_, JK_,
      nullptr, P(P_ATTN_A), scores);
  k_pool<<<B_, 256, 0, stream>>>(scores, jk, P(P_ATTN_GATE), alpha, graw);

  // ---------------- head
  k_small_mm<<<g1(B_ * FUS_, 256), 256, 0, stream>>>(graw, JK_, P(P_GP_W), P(P_GP_B),
                                                     graphL, FUS_, B_);
  k_small_bn<<<g1(FUS_, 128), 128, 0, stream>>>(graphL, B_, FUS_, P(P_GP_BN_G), P(P_GP_BN_B), 1);

  k_small_mm<<<g1(B_ * FUS_, 256), 256, 0, stream>>>(hc, FHC_, P(P_FE_W1), P(P_FE_B1),
                                                     f1, FUS_, B_);
  k_small_bn<<<g1(FUS_, 128), 128, 0, stream>>>(f1, B_, FUS_, P(P_FE_BN1_G), P(P_FE_BN1_B), 1);
  k_small_mm<<<g1(B_ * FUS_, 256), 256, 0, stream>>>(f1, FUS_, P(P_FE_W2), P(P_FE_B2),
                                                     f2, FUS_, B_);
  k_small_bn<<<g1(FUS_, 128), 128, 0, stream>>>(f2, B_, FUS_, P(P_FE_BN2_G), P(P_FE_BN2_B), 1);

  k_small_mm_cat<<<g1(B_ * H_, 256), 256, 0, stream>>>(graphL, FUS_, f2, FUS_,
                                                       P(P_CLS_W1), P(P_CLS_B1), xcls, H_, B_);
  k_small_bn<<<g1(H_, 128), 128, 0, stream>>>(xcls, B_, H_, P(P_CLS_BN_G), P(P_CLS_BN_B), 1);
  k_small_mm<<<g1(B_ * NC_, 256), 256, 0, stream>>>(xcls, H_, P(P_CLS_W2), P(P_CLS_B2),
                                                    (float*)d_out, NC_, B_);
}
